// RSSM_36867999269240
// MI455X (gfx1250) — matrix operand order repaired and checked by reading, not compiled, not run on hardware
//
#include <hip/hip_runtime.h>

typedef _Float16     v16h __attribute__((ext_vector_type(16)));
typedef float        v8f  __attribute__((ext_vector_type(8)));
typedef int          v4i  __attribute__((vector_size(16)));   // async-LDS builtin params
typedef unsigned int v4u  __attribute__((ext_vector_type(4))); // 16B copies (plain vector,
                                                               // works in any addrspace)

// Explicit LDS (addrspace(3)) pointer types so tile accesses lower to ds_* ops
// (generic pointers selected at runtime otherwise fall back to flat_*).
typedef __attribute__((address_space(3))) _Float16 lds_h;
typedef __attribute__((address_space(3))) v4u      lds_v4u;

#define B_    8192
#define NC_   32
#define CD_   32
#define AD_   128
#define H_    4096
#define BINS_ 255
#define BLK_  128
#define XDIM_ 1152     // NC*CD + AD
#define G3H_  12288    // 3*H
#define PDIM_ 5120     // H + NC*CD

#define LDS_STRIDE 40  // halfs per LDS row (32 + 8 pad) = 80B, 16B-aligned
#define CT_STRIDE  132 // floats per C-tile row
#define TILE_HALFS (128 * LDS_STRIDE)

// Generic LDS pointer -> addrspace(3): inttoptr to a 32-bit p3 pointer keeps
// the low 32 bits, which are the wave-relative LDS byte offset.
#define TO_LDS(p) ((lds_h*)(unsigned long long)(p))

// CDNA5 async global->LDS copy path (ASYNCcnt-tracked), if the toolchain
// declares the gfx1250 builtins; otherwise fall back to load+ds_store.
#if defined(__has_builtin)
#if __has_builtin(__builtin_amdgcn_global_load_async_to_lds_b128)
#define HAVE_ASYNC_LDS 1
#endif
#endif
#ifndef HAVE_ASYNC_LDS
#define HAVE_ASYNC_LDS 0
#endif

#if HAVE_ASYNC_LDS
#define AS1_V4I(p) ((__attribute__((address_space(1))) v4i*)(unsigned long long)(p))
#define AS3_V4I(p) ((__attribute__((address_space(3))) v4i*)(unsigned long long)(p))
#endif

// ---------------------------------------------------------------------------
// f32 -> f16 conversion with optional row re-stride (for concatenation).
// ---------------------------------------------------------------------------
__global__ void cvt_kernel(const float* __restrict__ src, _Float16* __restrict__ dst,
                           long long total, int rowlen, int dstld, int dstoff) {
    long long idx = (long long)blockIdx.x * blockDim.x + threadIdx.x;
    if (idx >= total) return;
    float v = src[idx];
    if (rowlen == dstld && dstoff == 0) {
        dst[idx] = (_Float16)v;
    } else {
        long long row = idx / rowlen;
        int col = (int)(idx - row * (long long)rowlen);
        dst[row * (long long)dstld + dstoff + col] = (_Float16)v;
    }
}

// ---------------------------------------------------------------------------
// Fence between staging and compute: drain ASYNCcnt (if async path in use)
// then workgroup barrier.
// ---------------------------------------------------------------------------
__device__ __forceinline__ void stage_fence() {
#if HAVE_ASYNC_LDS
#if __has_builtin(__builtin_amdgcn_s_wait_asynccnt)
    __builtin_amdgcn_s_wait_asynccnt(0);
#else
    asm volatile("s_wait_asynccnt 0x0" ::: "memory");
#endif
#endif
    __syncthreads();
}

// ---------------------------------------------------------------------------
// Stage one 128x32 A tile (row-major) and one 32x128 B tile (transposed to
// lB[n][k]) into LDS. A is assumed tile-complete in M (all launches use
// M = 8192, a multiple of 128). BVEC: B rows 16B-aligned and N-tile full.
// ---------------------------------------------------------------------------
template <bool BVEC>
__device__ __forceinline__ void stage_tile(
    const _Float16* __restrict__ A, int lda,
    const _Float16* __restrict__ Bm, int ldb,
    int N, int n0, int k0, int K,
    lds_h* lA, lds_h* lB) {
    const int tid = threadIdx.x;
    // ---- A tile: 128 rows x 32 halves; 32 bytes per thread ----
    {
        const int r   = tid >> 1;
        const int seg = (tid & 1) << 4;  // 0 or 16 halves
        const _Float16* gp = A + (size_t)r * lda + k0 + seg;
        lds_h* lp = lA + r * LDS_STRIDE + seg;
#if HAVE_ASYNC_LDS
        __builtin_amdgcn_global_load_async_to_lds_b128(AS1_V4I(gp), AS3_V4I(lp), 0, 0);
        __builtin_amdgcn_global_load_async_to_lds_b128(AS1_V4I(gp + 8), AS3_V4I(lp + 8), 0, 0);
#else
        const v4u* g4 = (const v4u*)gp;
        v4u d0 = g4[0], d1 = g4[1];
        lds_v4u* s4 = (lds_v4u*)lp;
        s4[0] = d0;
        s4[1] = d1;
#endif
    }
    // ---- B tile transposed: lB[n][k] ----
    {
        const int kk = tid >> 3;         // 0..31
        const int nb = (tid & 7) << 4;   // 0..112
        const _Float16* gp = Bm + (size_t)(k0 + kk) * ldb + n0 + nb;
        if (k0 + 32 < K)
            __builtin_prefetch(gp + (size_t)32 * ldb, 0, 1);
        if (BVEC) {
            union { v4u u; _Float16 h[8]; } d0, d1;
            d0.u = ((const v4u*)gp)[0];
            d1.u = ((const v4u*)gp)[1];
#pragma unroll
            for (int i = 0; i < 8; ++i) lB[(nb + i) * LDS_STRIDE + kk] = d0.h[i];
#pragma unroll
            for (int i = 0; i < 8; ++i) lB[(nb + 8 + i) * LDS_STRIDE + kk] = d1.h[i];
        } else {
#pragma unroll
            for (int i = 0; i < 16; ++i) {
                _Float16 v = (_Float16)0.f;
                if (n0 + nb + i < N) v = gp[i];
                lB[(nb + i) * LDS_STRIDE + kk] = v;
            }
        }
    }
}

// ---------------------------------------------------------------------------
// One K-slab of WMMAs for this wave's 64x32 sub-tile (4x2 fragments).
// Fragment loads match the CDNA5 16-bit A/B VGPR layouts:
//   A: lanes 0-15 K={0..7,16..23}, lanes 16-31 K={8..15,24..31}
//   B: lanes 0-15 K={0..7,16..23} of col, lanes 16-31 K={8..15,24..31}
// ---------------------------------------------------------------------------
__device__ __forceinline__ void compute_tile(const lds_h* lA, const lds_h* lB,
                                             v8f acc[4][2]) {
    const int tid  = threadIdx.x;
    const int lane = tid & 31;
    const int wave = tid >> 5;
    const int wmo  = (wave & 1) * 64;
    const int wno  = (wave >> 1) * 32;
    const int hl   = lane & 15;
    const int hk   = lane >> 4;

    v16h af[4], bf[2];
#pragma unroll
    for (int sm = 0; sm < 4; ++sm) {
        union { v4u u[2]; v16h h; } t;
        const lds_h* p = lA + (wmo + sm * 16 + hl) * LDS_STRIDE + hk * 8;
        t.u[0] = *(const lds_v4u*)p;         // K = kb .. kb+7
        t.u[1] = *(const lds_v4u*)(p + 16);  // K = kb+16 .. kb+23
        af[sm] = t.h;
    }
#pragma unroll
    for (int sn = 0; sn < 2; ++sn) {
        union { v4u u[2]; v16h h; } t;
        const lds_h* p = lB + (wno + sn * 16 + hl) * LDS_STRIDE + hk * 8;
        t.u[0] = *(const lds_v4u*)p;         // K = kb .. kb+7
        t.u[1] = *(const lds_v4u*)(p + 16);  // K = kb+16 .. kb+23
        bf[sn] = t.h;
    }
#pragma unroll
    for (int sm = 0; sm < 4; ++sm)
#pragma unroll
        for (int sn = 0; sn < 2; ++sn)
            acc[sm][sn] = __builtin_amdgcn_wmma_f32_16x16x32_f16(
                false, af[sm], false, bf[sn], (short)0, acc[sm][sn], false, false);
}

// ---------------------------------------------------------------------------
// Double-buffered GEMM core. A is pre-offset by m0*lda by the caller.
// Buffer selection is done with integer offsets on addrspace(3) pointers
// (no pointer arrays) so LDS accesses keep lowering to ds_* instructions.
// ---------------------------------------------------------------------------
template <bool BVEC>
__device__ __forceinline__ void wmma_core(
    const _Float16* __restrict__ A, int lda,
    const _Float16* __restrict__ Bm, int ldb,
    int N, int K, int n0,
    lds_h* lAbuf, lds_h* lBbuf,  // each 2 * TILE_HALFS
    v8f acc[4][2]) {
    stage_tile<BVEC>(A, lda, Bm, ldb, N, n0, 0, K, lAbuf, lBbuf);
    stage_fence();
    int cur = 0;
    for (int k0 = 0; k0 < K; k0 += 32) {
        const int co = cur ? TILE_HALFS : 0;
        const int no = cur ? 0 : TILE_HALFS;
        if (k0 + 32 < K)
            stage_tile<BVEC>(A, lda, Bm, ldb, N, n0, k0 + 32, K, lAbuf + no, lBbuf + no);
        compute_tile(lAbuf + co, lBbuf + co, acc);
        stage_fence();
        cur ^= 1;
    }
}

// ---------------------------------------------------------------------------
// EPI 0: out[m*ldc+n] = acc + bias[n]            (gx GEMM, r/v/c heads)
// EPI 1: block-diag gh fold: gate g = n>>7 of hidden block bz.
//        g<2  -> gates[m][g*H + bz*128 + k] += val   (r,z gate pre-sums)
//        g==2 -> ghn[m][bz*128 + k] = val            (hn kept separate)
// ---------------------------------------------------------------------------
template <int EPI, bool BVEC>
__global__ void wmma_gemm_kernel(
    const _Float16* __restrict__ Abase, int lda,
    const _Float16* __restrict__ Bbase, int ldb,
    const float* __restrict__ biasbase,
    int M, int N, int K,
    float* __restrict__ out, int ldc,
    float* __restrict__ gates, float* __restrict__ ghn) {
    __shared__ _Float16 lA[2 * TILE_HALFS];
    __shared__ _Float16 lB[2 * TILE_HALFS];
    lds_h* sA = TO_LDS(lA);
    lds_h* sB = TO_LDS(lB);
    const int m0 = blockIdx.y * 128;
    const int n0 = blockIdx.x * 128;

    const _Float16* A  = Abase;
    const _Float16* Bm = Bbase;
    const float* bias  = biasbase;
    int bz = 0;
    if (EPI == 1) {
        bz   = blockIdx.z;
        A    = Abase + bz * BLK_;                 // h_prev block columns, lda = H
        Bm   = Bbase + (size_t)bz * BLK_ * 384;   // W_hh[bz] : 128 x 384
        bias = biasbase + bz * 384;               // b_hh[bz]
    }
    A += (size_t)m0 * lda;

    v8f zero = {0.f, 0.f, 0.f, 0.f, 0.f, 0.f, 0.f, 0.f};
    v8f acc[4][2];
#pragma unroll
    for (int i = 0; i < 4; ++i)
#pragma unroll
        for (int j = 0; j < 2; ++j) acc[i][j] = zero;

    wmma_core<BVEC>(A, lda, Bm, ldb, N, K, n0, sA, sB, acc);

    const int tid = threadIdx.x;
    const int lane = tid & 31, wave = tid >> 5;
    const int wmo = (wave & 1) * 64, wno = (wave >> 1) * 32;
    const int hl = lane & 15, hk = lane >> 4;
#pragma unroll
    for (int sm = 0; sm < 4; ++sm)
#pragma unroll
        for (int sn = 0; sn < 2; ++sn)
#pragma unroll
            for (int v = 0; v < 8; ++v) {
                int m = m0 + wmo + sm * 16 + v + hk * 8;
                int n = n0 + wno + sn * 16 + hl;
                if (m < M && n < N) {
                    float val = acc[sm][sn][v] + bias[n];
                    if (EPI == 0) {
                        out[(size_t)m * ldc + n] = val;
                    } else {
                        int g = n >> 7, kk = n & 127;
                        if (g < 2)
                            gates[(size_t)m * G3H_ + g * H_ + bz * BLK_ + kk] += val;
                        else
                            ghn[(size_t)m * H_ + bz * BLK_ + kk] = val;
                    }
                }
            }
}

// ---------------------------------------------------------------------------
// logits GEMM (N=1024) with unimix epilogue; optional gumbel-hard z sampling.
// Softmax groups are 32 columns (one categorical), 4 groups per 128-wide tile.
// ---------------------------------------------------------------------------
template <bool GUMBEL>
__global__ void wmma_gemm_unimix_kernel(
    const _Float16* __restrict__ A, int lda,
    const _Float16* __restrict__ Bm, int ldb,
    const float* __restrict__ bias, int M, int K,
    float* __restrict__ outlog,
    const float* __restrict__ gum,
    float* __restrict__ zout,
    _Float16* __restrict__ feat16) {
    __shared__ _Float16 lA[2 * TILE_HALFS];
    __shared__ _Float16 lB[2 * TILE_HALFS];
    __shared__ float ct[128 * CT_STRIDE];
    lds_h* sA = TO_LDS(lA);
    lds_h* sB = TO_LDS(lB);
    const int m0 = blockIdx.y * 128;
    const int n0 = blockIdx.x * 128;

    v8f zero = {0.f, 0.f, 0.f, 0.f, 0.f, 0.f, 0.f, 0.f};
    v8f acc[4][2];
#pragma unroll
    for (int i = 0; i < 4; ++i)
#pragma unroll
        for (int j = 0; j < 2; ++j) acc[i][j] = zero;

    wmma_core<true>(A + (size_t)m0 * lda, lda, Bm, ldb, 1024, K, n0, sA, sB, acc);

    const int tid = threadIdx.x;
    const int lane = tid & 31, wave = tid >> 5;
    const int wmo = (wave & 1) * 64, wno = (wave >> 1) * 32;
    const int hl = lane & 15, hk = lane >> 4;
#pragma unroll
    for (int sm = 0; sm < 4; ++sm)
#pragma unroll
        for (int sn = 0; sn < 2; ++sn)
#pragma unroll
            for (int v = 0; v < 8; ++v) {
                int rl = wmo + sm * 16 + v + hk * 8;
                int cl = wno + sn * 16 + hl;
                ct[rl * CT_STRIDE + cl] = acc[sm][sn][v] + bias[n0 + cl];
            }
    __syncthreads();

    // 128 rows x 4 groups of 32 = 512 groups; 2 per thread.
    for (int g = tid; g < 512; g += 256) {
        int row = g >> 2;
        int gq  = g & 3;
        const float* lr = ct + row * CT_STRIDE + gq * 32;
        float mx = lr[0];
#pragma unroll
        for (int i = 1; i < 32; ++i) mx = fmaxf(mx, lr[i]);
        float se = 0.f;
#pragma unroll
        for (int i = 0; i < 32; ++i) se += expf(lr[i] - mx);
        float inv = 1.f / se;
        size_t ob = (size_t)(m0 + row) * 1024 + n0 + gq * 32;
        float lp[32];
#pragma unroll
        for (int i = 0; i < 32; ++i) {
            float p = expf(lr[i] - mx) * inv;
            lp[i]   = logf(0.99f * p + 0.0003125f);  // (1-a)*p + a/CD
            outlog[ob + i] = lp[i];
        }
        if (GUMBEL) {
            int amax = 0;
            float best = -1e30f;
#pragma unroll
            for (int i = 0; i < 32; ++i) {
                float u  = gum[ob + i];
                float gn = -logf(-logf(u + 1e-9f) + 1e-9f);
                float t  = lp[i] + gn;
                if (t > best) { best = t; amax = i; }
            }
            size_t fb = (size_t)(m0 + row) * PDIM_ + H_ + n0 + gq * 32;
#pragma unroll
            for (int i = 0; i < 32; ++i) {
                float zv = (i == amax) ? 1.f : 0.f;
                zout[ob + i]  = zv;
                feat16[fb + i] = (_Float16)zv;  // forward value of straight-through
            }
        }
    }
}

// ---------------------------------------------------------------------------
// GRU elementwise + RMSNorm + SiLU. One workgroup per batch row.
// ---------------------------------------------------------------------------
__global__ void gru_norm_kernel(const float* __restrict__ gates,
                                const float* __restrict__ ghn,
                                const float* __restrict__ h_prev,
                                const float* __restrict__ norm_w,
                                float* __restrict__ out_h,
                                _Float16* __restrict__ post16,
                                _Float16* __restrict__ feat16) {
    const int b = blockIdx.x;
    const int tid = threadIdx.x;
    __shared__ float red[256];
    const size_t gbase = (size_t)b * G3H_;
    const size_t hbase = (size_t)b * H_;

    float hraw[16];
    float ss = 0.f;
#pragma unroll
    for (int i = 0; i < 16; ++i) {
        int j = i * 256 + tid;
        float xr = gates[gbase + j];            // gx_r + gh_r (+ both biases)
        float xz = gates[gbase + H_ + j];       // gx_z + gh_z
        float xn = gates[gbase + 2 * H_ + j];   // gx_n only
        float hn = ghn[hbase + j];              // gh_n (+ b_hh)
        float r  = 1.f / (1.f + expf(-xr));
        float u  = 1.f / (1.f + expf(-xz));
        float nv = tanhf(xn + r * hn);
        float hv = (1.f - u) * nv + u * h_prev[hbase + j];
        hraw[i]  = hv;
        ss += hv * hv;
    }
    red[tid] = ss;
    __syncthreads();
    for (int s = 128; s > 0; s >>= 1) {
        if (tid < s) red[tid] += red[tid + s];
        __syncthreads();
    }
    float scale = rsqrtf(red[0] / (float)H_ + 1e-6f);
#pragma unroll
    for (int i = 0; i < 16; ++i) {
        int j = i * 256 + tid;
        float hn2 = hraw[i] * scale * norm_w[j];
        float sv  = hn2 / (1.f + expf(-hn2));   // silu
        out_h[hbase + j] = sv;
        _Float16 sh = (_Float16)sv;
        post16[(size_t)b * PDIM_ + j] = sh;
        feat16[(size_t)b * PDIM_ + j] = sh;
    }
}

// ---------------------------------------------------------------------------
// Launch
// ---------------------------------------------------------------------------
extern "C" void kernel_launch(void* const* d_in, const int* in_sizes, int n_in,
                              void* d_out, int out_size, void* d_ws, size_t ws_size,
                              hipStream_t stream) {
    (void)in_sizes; (void)n_in; (void)out_size; (void)ws_size;

    const float* h_prev  = (const float*)d_in[0];
    const float* z_prev  = (const float*)d_in[1];
    const float* action  = (const float*)d_in[2];
    const float* embed   = (const float*)d_in[3];
    const float* gum     = (const float*)d_in[4];
    const float* W_ih    = (const float*)d_in[5];
    const float* b_ih    = (const float*)d_in[6];
    const float* W_hh    = (const float*)d_in[7];
    const float* b_hh    = (const float*)d_in[8];
    const float* norm_w  = (const float*)d_in[9];
    const float* W_prior = (const float*)d_in[10];
    const float* b_prior = (const float*)d_in[11];
    const float* W_post  = (const float*)d_in[12];
    const float* b_post  = (const float*)d_in[13];
    const float* W_r     = (const float*)d_in[14];
    const float* b_r     = (const float*)d_in[15];
    const float* W_v     = (const float*)d_in[16];
    const float* b_v     = (const float*)d_in[17];
    const float* W_c     = (const float*)d_in[18];
    const float* b_c     = (const float*)d_in[19];

    float* out   = (float*)d_out;
    float* out_h = out;
    float* out_z = out_h + (size_t)B_ * H_;
    float* out_p = out_z + (size_t)B_ * NC_ * CD_;
    float* out_q = out_p + (size_t)B_ * NC_ * CD_;
    float* out_r = out_q + (size_t)B_ * NC_ * CD_;
    float* out_v = out_r + (size_t)B_ * BINS_;
    float* out_c = out_v + (size_t)B_ * BINS_;

    // workspace carve-up
    char* w = (char*)d_ws;
    size_t off = 0;
    auto take = [&](size_t bytes) -> void* {
        void* p = w + off;
        off = (off + bytes + 255) & ~(size_t)255;
        return p;
    };
    _Float16* xh     = (_Float16*)take((size_t)B_ * XDIM_ * 2);
    _Float16* Wih16  = (_Float16*)take((size_t)XDIM_ * G3H_ * 2);
    _Float16* hp16   = (_Float16*)take((size_t)B_ * H_ * 2);
    _Float16* Whh16  = (_Float16*)take((size_t)NC_ * BLK_ * 384 * 2);
    _Float16* Wpr16  = (_Float16*)take((size_t)H_ * 1024 * 2);
    _Float16* Wpo16  = (_Float16*)take((size_t)PDIM_ * 1024 * 2);
    _Float16* Wr16   = (_Float16*)take((size_t)PDIM_ * BINS_ * 2);
    _Float16* Wv16   = (_Float16*)take((size_t)PDIM_ * BINS_ * 2);
    _Float16* Wc16   = (_Float16*)take((size_t)PDIM_ * 2);
    float*    gates  = (float*)take((size_t)B_ * G3H_ * 4);
    float*    ghn    = (float*)take((size_t)B_ * H_ * 4);
    _Float16* post16 = (_Float16*)take((size_t)B_ * PDIM_ * 2);  // h || embed_obs
    _Float16* feat16 = (_Float16*)take((size_t)B_ * PDIM_ * 2);  // h || z

    auto cvt = [&](const float* s, _Float16* d, long long total, int rowlen,
                   int dld, int doff) {
        int blocks = (int)((total + 255) / 256);
        cvt_kernel<<<blocks, 256, 0, stream>>>(s, d, total, rowlen, dld, doff);
    };
    // activations (with concat re-stride)
    cvt(z_prev, xh,     (long long)B_ * NC_ * CD_, NC_ * CD_, XDIM_, 0);
    cvt(action, xh,     (long long)B_ * AD_,       AD_,       XDIM_, NC_ * CD_);
    cvt(h_prev, hp16,   (long long)B_ * H_,        H_,        H_,    0);
    cvt(embed,  post16, (long long)B_ * NC_ * CD_, NC_ * CD_, PDIM_, H_);
    // weights
    cvt(W_ih,    Wih16, (long long)XDIM_ * G3H_,      G3H_,  G3H_,  0);
    cvt(W_hh,    Whh16, (long long)NC_ * BLK_ * 384,  384,   384,   0);
    cvt(W_prior, Wpr16, (long long)H_ * 1024,         1024,  1024,  0);
    cvt(W_post,  Wpo16, (long long)PDIM_ * 1024,      1024,  1024,  0);
    cvt(W_r,     Wr16,  (long long)PDIM_ * BINS_,     BINS_, BINS_, 0);
    cvt(W_v,     Wv16,  (long long)PDIM_ * BINS_,     BINS_, BINS_, 0);
    cvt(W_c,     Wc16,  (long long)PDIM_,             1,     1,     0);

    // 1) gx = x @ W_ih + b_ih            [8192 x 12288], K=1152
    wmma_gemm_kernel<0, true><<<dim3(G3H_ / 128, B_ / 128), 256, 0, stream>>>(
        xh, XDIM_, Wih16, G3H_, b_ih, B_, G3H_, XDIM_, gates, G3H_, nullptr, nullptr);

    // 2) gh fold: 32 blocks of [8192x128]@[128x384], r/z summed into gates,
    //    n-gate hn kept in ghn
    wmma_gemm_kernel<1, true><<<dim3(3, B_ / 128, NC_), 256, 0, stream>>>(
        hp16, H_, Whh16, 384, b_hh, B_, 384, BLK_, nullptr, 0, gates, ghn);

    // 3) GRU + RMSNorm + SiLU -> h (f32 out) + f16 into post/feat staging
    gru_norm_kernel<<<B_, 256, 0, stream>>>(gates, ghn, h_prev, norm_w,
                                            out_h, post16, feat16);

    // 4) prior logits: h @ W_prior + b_prior -> unimix log-probs
    wmma_gemm_unimix_kernel<false><<<dim3(1024 / 128, B_ / 128), 256, 0, stream>>>(
        post16, PDIM_, Wpr16, 1024, b_prior, B_, H_, out_p, nullptr, nullptr, nullptr);

    // 5) posterior logits: [h||embed] @ W_post + b_post -> unimix + gumbel z
    wmma_gemm_unimix_kernel<true><<<dim3(1024 / 128, B_ / 128), 256, 0, stream>>>(
        post16, PDIM_, Wpo16, 1024, b_post, B_, PDIM_, out_q, gum, out_z, feat16);

    // 6) heads: feat @ {W_r, W_v, W_c}
    wmma_gemm_kernel<0, false><<<dim3(2, B_ / 128), 256, 0, stream>>>(
        feat16, PDIM_, Wr16, BINS_, b_r, B_, BINS_, PDIM_, out_r, BINS_, nullptr, nullptr);
    wmma_gemm_kernel<0, false><<<dim3(2, B_ / 128), 256, 0, stream>>>(
        feat16, PDIM_, Wv16, BINS_, b_v, B_, BINS_, PDIM_, out_v, BINS_, nullptr, nullptr);
    wmma_gemm_kernel<0, false><<<dim3(1, B_ / 128), 256, 0, stream>>>(
        feat16, PDIM_, Wc16, 1, b_c, B_, 1, PDIM_, out_c, 1, nullptr, nullptr);
}
